// Discriminator_16509854285886
// MI455X (gfx1250) — matrix operand order repaired and checked by reading, not compiled, not run on hardware
//
#include <hip/hip_runtime.h>
#include <hip/hip_bf16.h>

typedef __bf16 v16bf __attribute__((ext_vector_type(16)));
typedef __bf16 v8bf  __attribute__((ext_vector_type(8)));
typedef __bf16 v4bf  __attribute__((ext_vector_type(4)));
typedef float  v8f   __attribute__((ext_vector_type(8)));
typedef float  v4f   __attribute__((ext_vector_type(4)));

#define BM 64
#define BN 128
#define BK 32
#define LDSK (BK + 8)   // 40 bf16 = 80B row stride: 16B-aligned, bank-staggered

union FragU { v16bf v; v8bf h[2]; };

// C[M x N] = A[M x K] * B[N x K]^T  (+ bias[col]) (+ addmat[row][col])
// A, B fp32 K-contiguous (NT GEMM); converted to bf16 while staging into LDS.
// Double-buffered LDS, one barrier per k-tile, last iteration peeled.
// Requires M % 64 == 0, N % 128 == 0, K % 32 == 0.
template<bool HAS_BIAS, bool HAS_ADD>
__global__ __launch_bounds__(256)
void gemm_nt_bf16(const float* __restrict__ A, int lda,
                  const float* __restrict__ B, int ldb,
                  float* __restrict__ C, int ldc,
                  const float* __restrict__ addmat, int ldadd,
                  const float* __restrict__ bias,
                  int K)
{
    __shared__ __align__(16) __bf16 As[2][BM][LDSK];
    __shared__ __align__(16) __bf16 Bs[2][BN][LDSK];

    const int tid  = threadIdx.x;
    const int wave = tid >> 5;
    const int lane = tid & 31;
    const int half = lane >> 4;        // 0: lanes 0-15, 1: lanes 16-31
    const int l16  = lane & 15;
    const int wm   = (wave >> 2) * 32; // wave row offset in block tile
    const int wn   = (wave & 3) * 32;  // wave col offset
    const int m0   = blockIdx.x * BM;
    const int n0   = blockIdx.y * BN;

    // staging geometry: each thread moves 2 A-float4s + 4 B-float4s per tile
    const int arow0 = tid >> 3;        // 0..31
    const int arow1 = arow0 + 32;
    const int acol  = (tid & 7) << 2;  // 0,4,...,28

    const float* Ab = A + (size_t)m0 * lda + acol;
    const float* Bb = B + (size_t)n0 * ldb + acol;

    v4f ra[2], rb[4];
    v8f acc[2][2] = {};

    const int ktiles = K / BK;

    auto global_fetch = [&](int kt) {
        const int kb = kt * BK;
        ra[0] = *(const v4f*)&Ab[(size_t)arow0 * lda + kb];
        ra[1] = *(const v4f*)&Ab[(size_t)arow1 * lda + kb];
#pragma unroll
        for (int i = 0; i < 4; ++i)
            rb[i] = *(const v4f*)&Bb[(size_t)(i * 32 + arow0) * ldb + kb];
    };
    auto lds_store = [&](int buf) {
        *(v4bf*)&As[buf][arow0][acol] = __builtin_convertvector(ra[0], v4bf);
        *(v4bf*)&As[buf][arow1][acol] = __builtin_convertvector(ra[1], v4bf);
#pragma unroll
        for (int i = 0; i < 4; ++i)
            *(v4bf*)&Bs[buf][i * 32 + arow0][acol] = __builtin_convertvector(rb[i], v4bf);
    };
    // fragments per ISA VGPR layout (wave32, 16x16x32 bf16) + 4 WMMAs
    auto compute = [&](int buf) {
        FragU af[2], bfr[2];
#pragma unroll
        for (int am = 0; am < 2; ++am) {
            int row = wm + am * 16 + l16;            // A: M = lane%16
            // lanes<16: e0-7=K0..7, e8-15=K16..23 ; lanes>=16: e0-7=K8..15, e8-15=K24..31
            af[am].h[0] = *(const v8bf*)&As[buf][row][half * 8];
            af[am].h[1] = *(const v8bf*)&As[buf][row][16 + half * 8];
        }
#pragma unroll
        for (int bn = 0; bn < 2; ++bn) {
            int colv = wn + bn * 16 + l16;           // B: N = lane%16
            bfr[bn].h[0] = *(const v8bf*)&Bs[buf][colv][half * 8];
            bfr[bn].h[1] = *(const v8bf*)&Bs[buf][colv][16 + half * 8];
        }
#pragma unroll
        for (int am = 0; am < 2; ++am)
#pragma unroll
            for (int bn = 0; bn < 2; ++bn)
                acc[am][bn] = __builtin_amdgcn_wmma_f32_16x16x32_bf16(
                    false, af[am].v, false, bfr[bn].v,
                    (short)0, acc[am][bn], false, false);
    };

    global_fetch(0);
    lds_store(0);

    // steady state: unconditional fetch(kt+1) / compute(kt) / store(kt+1)
    for (int kt = 0; kt < ktiles - 1; ++kt) {
        __syncthreads();
        const int buf = kt & 1;
        global_fetch(kt + 1);
        if (kt + 2 < ktiles) {  // L2 prefetch two tiles ahead (global_prefetch_b8)
            __builtin_prefetch(&Ab[(size_t)arow0 * lda + (kt + 2) * BK], 0, 1);
            __builtin_prefetch(&Bb[(size_t)arow0 * ldb + (kt + 2) * BK], 0, 1);
        }
        compute(buf);
        lds_store(buf ^ 1);
    }
    __syncthreads();
    compute((ktiles - 1) & 1);   // peeled last tile

    // ---- branch-free epilogue: lane -> N = lane%16, M = half*8 + vgpr index
#pragma unroll
    for (int am = 0; am < 2; ++am) {
        const int rbase = m0 + wm + am * 16 + half * 8;
#pragma unroll
        for (int bn = 0; bn < 2; ++bn) {
            const int colg = n0 + wn + bn * 16 + l16;
            float* cp = C + (size_t)rbase * ldc + colg;
            const float* ap = HAS_ADD ? (addmat + (size_t)rbase * ldadd + colg)
                                      : nullptr;
            float bia;
            if constexpr (HAS_BIAS) bia = bias[colg];
#pragma unroll
            for (int i = 0; i < 8; ++i) {
                float val = acc[am][bn][i];
                if constexpr (HAS_BIAS) val += bia;
                if constexpr (HAS_ADD)  val += ap[(size_t)i * ldadd];
                cp[(size_t)i * ldc] = val;
            }
        }
    }
}

__global__ void zero_kernel(float* __restrict__ p, int n) {
    int i = blockIdx.x * 256 + threadIdx.x;
    if (i < n) p[i] = 0.f;
}

// vt[:, 512:640] = t[b,l] * Wt_up_w[j] + Wt_up_b[j]
__global__ void te_fill(const float* __restrict__ t, const float* __restrict__ w,
                        const float* __restrict__ b, float* __restrict__ vt) {
    int idx = blockIdx.x * 256 + threadIdx.x;   // 2048*128
    if (idx >= 2048 * 128) return;
    int j = idx & 127, bl = idx >> 7;
    vt[(size_t)bl * 640 + 512 + j] = t[bl] * w[j] + b[j];
}

__device__ __forceinline__ float sigf(float x) { return 1.f / (1.f + __expf(-x)); }

__global__ void lstm_elem(const float* __restrict__ gates,
                          float* __restrict__ h, float* __restrict__ c) {
    int idx = blockIdx.x * 256 + threadIdx.x;   // 128*1024
    if (idx >= 128 * 1024) return;
    int b = idx >> 10, j = idx & 1023;
    const float* g = gates + (size_t)b * 4096;
    float ig = sigf(g[j]);
    float fg = sigf(g[1024 + j]);
    float gg = tanhf(g[2048 + j]);
    float og = sigf(g[3072 + j]);
    float cn = fg * c[idx] + ig * gg;
    c[idx] = cn;
    h[idx] = og * tanhf(cn);
}

// out[b] = dot(h[b,:], lin_w) + lin_b ; one wave32 per row
__global__ void pred_kernel(const float* __restrict__ h, const float* __restrict__ lw,
                            const float* __restrict__ lb, float* __restrict__ out) {
    int b = blockIdx.x;
    int lane = threadIdx.x;
    float s = 0.f;
    for (int k = lane; k < 1024; k += 32) s += h[(size_t)b * 1024 + k] * lw[k];
#pragma unroll
    for (int off = 16; off > 0; off >>= 1) s += __shfl_down(s, off, 32);
    if (lane == 0) out[b] = s + lb[0];
}

extern "C" void kernel_launch(void* const* d_in, const int* in_sizes, int n_in,
                              void* d_out, int out_size, void* d_ws, size_t ws_size,
                              hipStream_t stream) {
    const float* v       = (const float*)d_in[0];   // [128,16,20000]
    const float* t       = (const float*)d_in[1];   // [128,16]
    const float* W_down  = (const float*)d_in[2];   // [512,20000]
    const float* Wt_up_w = (const float*)d_in[3];   // [128,1]
    const float* Wt_up_b = (const float*)d_in[4];   // [128]
    const float* W_vt    = (const float*)d_in[5];   // [512,640]
    const float* W_ih    = (const float*)d_in[6];   // [4096,512]
    const float* W_hh    = (const float*)d_in[7];   // [4096,1024]
    const float* b_ih    = (const float*)d_in[8];   // [4096]
    const float* b_hh    = (const float*)d_in[9];   // [4096]
    const float* lin_w   = (const float*)d_in[10];  // [1,1024]
    const float* lin_b   = (const float*)d_in[11];  // [1]
    float* out = (float*)d_out;

    float* ws    = (float*)d_ws;
    float* vt    = ws;                       // 2048*640
    float* inp   = vt + 2048 * 640;          // 2048*512
    float* xg    = inp + 2048 * 512;         // 2048*4096
    float* gates = xg + 2048 * 4096;         // 128*4096
    float* h     = gates + 128 * 4096;       // 128*1024
    float* c     = h + 128 * 1024;           // 128*1024  (h,c contiguous)

    // h = c = 0
    zero_kernel<<<(128 * 1024 * 2 + 255) / 256, 256, 0, stream>>>(h, 128 * 1024 * 2);

    // vd = v @ W_down^T  -> vt[:, 0:512]   (M=2048,N=512,K=20000)
    gemm_nt_bf16<false, false><<<dim3(2048 / BM, 512 / BN), 256, 0, stream>>>(
        v, 20000, W_down, 20000, vt, 640, nullptr, 0, nullptr, 20000);

    // vt[:, 512:640] = t * Wt_up_w + Wt_up_b
    te_fill<<<(2048 * 128) / 256, 256, 0, stream>>>(t, Wt_up_w, Wt_up_b, vt);

    // inputs = vt @ W_vt^T                 (M=2048,N=512,K=640)
    gemm_nt_bf16<false, false><<<dim3(2048 / BM, 512 / BN), 256, 0, stream>>>(
        vt, 640, W_vt, 640, inp, 512, nullptr, 0, nullptr, 640);

    // xg = inputs @ W_ih^T + b_ih          (M=2048,N=4096,K=512)
    gemm_nt_bf16<true, false><<<dim3(2048 / BM, 4096 / BN), 256, 0, stream>>>(
        inp, 512, W_ih, 512, xg, 4096, nullptr, 0, b_ih, 512);

    // LSTM scan: gates = xg[:,l,:] + h @ W_hh^T + b_hh ; pointwise update
    for (int l = 0; l < 16; ++l) {
        gemm_nt_bf16<true, true><<<dim3(128 / BM, 4096 / BN), 256, 0, stream>>>(
            h, 1024, W_hh, 1024, gates, 4096,
            xg + (size_t)l * 4096, 16 * 4096, b_hh, 1024);
        lstm_elem<<<(128 * 1024) / 256, 256, 0, stream>>>(gates, h, c);
    }

    // pred = h @ lin_w^T + lin_b
    pred_kernel<<<128, 32, 0, stream>>>(h, lin_w, lin_b, out);
}
